// RoutingCapsule_28999619182845
// MI455X (gfx1250) — compile-verified, matrix operand order checked by reading, not run on hardware
//
#include <hip/hip_runtime.h>
#include <hip/hip_fp16.h>

typedef __attribute__((ext_vector_type(16))) _Float16 v16h;
typedef __attribute__((ext_vector_type(8)))  _Float16 v8h;
typedef __attribute__((ext_vector_type(8)))  float    v8f;

#define BB     32
#define QTOT   4608            // P*H*W
#define NPRIM  8
#define NCAP   16
#define DLEN   32
#define OC     512             // NCAP*DLEN
#define NQC    32
#define QCHUNK (QTOT / NQC)    // 144

// ---------------------------------------------------------------------------
// Kernel 1: u[b,q,o] = sum_i x[b,q,i] * W[q,i,o], stored fp16.
// One block (8 waves) per q. WMMA f16 16x16x32 with K padded 8 -> 32.
// A = W^T (M = o rows), B = x (N = b cols): the f32 D layout hands each lane
// 8 consecutive o values -> one packed 16B fp16 store, no epilogue transpose.
// LDS rows are padded to 16 halves with a zeroed upper half, so each operand
// is two ds_load_b128 directly into the WMMA register range; lanes >= 16
// (K = 8..15 / 24..31) read a 32B zero row via one address cndmask.
// ---------------------------------------------------------------------------
__global__ __launch_bounds__(256) void k_compute_u(const float* __restrict__ x,
                                                   const float* __restrict__ w,
                                                   _Float16* __restrict__ u) {
    __shared__ __align__(32) _Float16 xl[BB * 16];   // [b][0:8 data | 8:16 zero]
    __shared__ __align__(32) _Float16 wl[OC * 16];   // [col][0:8 data | 8:16 zero]
    __shared__ __align__(32) _Float16 zrow[16];      // all-zero operand row
    const int q   = blockIdx.x;
    const int tid = threadIdx.x;

    v8h z8;
#pragma unroll
    for (int e = 0; e < 8; ++e) z8[e] = (_Float16)0.0f;

    if (tid < 16) zrow[tid] = (_Float16)0.0f;
    if (tid < BB) *(v8h*)&xl[tid * 16 + 8] = z8;     // zero upper halves (x)
    *(v8h*)&wl[(2 * tid) * 16 + 8]     = z8;         // zero upper halves (W)
    *(v8h*)&wl[(2 * tid + 1) * 16 + 8] = z8;

    // stage x[:, q, :] (256 floats -> fp16)
    {
        int b = tid >> 3, i = tid & 7;
        xl[b * 16 + i] = (_Float16)x[((size_t)b * QTOT + q) * NPRIM + i];
    }
    // stage W[q] transposed (4096 floats -> fp16), coalesced global reads
    const float* wq = w + (size_t)q * (NPRIM * OC);
#pragma unroll
    for (int t = 0; t < 16; ++t) {
        int f = tid + t * 256;
        int i = f >> 9, col = f & (OC - 1);
        wl[col * 16 + i] = (_Float16)wq[f];
    }
    __syncthreads();

    const int wave = tid >> 5, lane = tid & 31;
    const int bt   = wave & 1;     // which 16 of the 32 batch cols
    const int og   = wave >> 1;    // o-group: 8 M-tiles of 16 each
    const bool lo  = (lane < 16);
    const int l15  = lane & 15;

    // B operand: x tile, 32(K)x16(N=b). Real K = 0..7.
    const _Float16* bsrc = lo ? &xl[(bt * 16 + l15) * 16] : zrow;
    const v16h bmat = *(const v16h*)bsrc;              // 2x ds_load_b128

#pragma unroll
    for (int ot = 0; ot < 8; ++ot) {
        const int obase = (og * 8 + ot) * 16;
        // A operand: W^T tile, 16(M=o)x32(K).
        const _Float16* asrc = lo ? &wl[(obase + l15) * 16] : zrow;
        const v16h amat = *(const v16h*)asrc;          // 2x ds_load_b128

        v8f acc = {};
        acc = __builtin_amdgcn_wmma_f32_16x16x32_f16(
                  false, amat, false, bmat, (short)0, acc, false, false);

        // D layout: lane -> N=b col (lane&15), VGPR r -> M = obase+8*(lane>>4)+r
        const int b = bt * 16 + l15;
        const int o = obase + ((lane >> 4) << 3);
        v8h outv;
#pragma unroll
        for (int r = 0; r < 8; ++r) outv[r] = (_Float16)acc[r];
        *(v8h*)&u[((size_t)b * QTOT + q) * OC + o] = outv;   // 16B packed store
    }
}

// ---------------------------------------------------------------------------
__global__ __launch_bounds__(256) void k_zero(float* __restrict__ p, int n) {
    int i = blockIdx.x * 256 + threadIdx.x;
    if (i < n) p[i] = 0.0f;
}

// softmax over n_cap for each (b,q)
__global__ __launch_bounds__(256) void k_softmax(const float* __restrict__ bias,
                                                 float* __restrict__ c) {
    int idx = blockIdx.x * 256 + threadIdx.x;   // flattened (b*Q + q)
    if (idx >= BB * QTOT) return;
    const float* bp = bias + (size_t)idx * NCAP;
    float vv[NCAP];
#pragma unroll
    for (int n = 0; n < NCAP; ++n) vv[n] = bp[n];
    float m = vv[0];
#pragma unroll
    for (int n = 1; n < NCAP; ++n) m = fmaxf(m, vv[n]);
    float s = 0.0f;
#pragma unroll
    for (int n = 0; n < NCAP; ++n) { vv[n] = __expf(vv[n] - m); s += vv[n]; }
    float inv = 1.0f / s;
    float* cp = c + (size_t)idx * NCAP;
#pragma unroll
    for (int n = 0; n < NCAP; ++n) cp[n] = vv[n] * inv;
}

// s_part[qc, b*512 + o] = sum_{q in chunk qc} c[b,q,o>>5] * u[b,q,o]
// wave = (b, o-half, q-chunk); lane covers 8 consecutive o (16B fp16 load),
// all within one capsule n, so one broadcast c scalar per lane per q.
// 2048 independent streaming waves keep HBM/L2 saturated.
__global__ __launch_bounds__(256) void k_route_s(const _Float16* __restrict__ u,
                                                 const float* __restrict__ c,
                                                 float* __restrict__ spart) {
    const int wave = threadIdx.x >> 5, lane = threadIdx.x & 31;
    const int item = blockIdx.x * 8 + wave;    // 0..2047
    const int qc = item & (NQC - 1);
    const int oq = (item >> 5) & 1;
    const int b  = item >> 6;
    const int o  = oq * 256 + lane * 8;        // 8 consecutive o, same n
    const int n  = o >> 5;
    const _Float16* up = u + (size_t)b * QTOT * OC + o;
    const float*    cp = c + (size_t)b * QTOT * NCAP + n;
    float a[8];
#pragma unroll
    for (int j = 0; j < 8; ++j) a[j] = 0.0f;
    const int q0 = qc * QCHUNK;
#pragma unroll 4
    for (int qq = 0; qq < QCHUNK; ++qq) {
        const size_t q = (size_t)(q0 + qq);
        v8h  uv  = *(const v8h*)(up + q * OC);
        float cc = cp[q * NCAP];
#pragma unroll
        for (int j = 0; j < 8; ++j) a[j] += cc * (float)uv[j];
    }
    float* dst = &spart[(size_t)qc * (BB * OC) + (size_t)b * OC + o];
    float4 r0; r0.x = a[0]; r0.y = a[1]; r0.z = a[2]; r0.w = a[3];
    float4 r1; r1.x = a[4]; r1.y = a[5]; r1.z = a[6]; r1.w = a[7];
    *(float4*)dst       = r0;
    *(float4*)(dst + 4) = r1;
}

// deterministic reduction of the 32 q-chunk partials
__global__ __launch_bounds__(256) void k_reduce_s(const float* __restrict__ spart,
                                                  float* __restrict__ s) {
    int i = blockIdx.x * 256 + threadIdx.x;    // < BB*OC = 16384
    float t = 0.0f;
#pragma unroll
    for (int qc = 0; qc < NQC; ++qc) t += spart[(size_t)qc * (BB * OC) + i];
    s[i] = t;
}

// squash: wave per (b,n), lane = d; writes v scratch and d_out (last iter wins)
__global__ __launch_bounds__(256) void k_squash(const float* __restrict__ s,
                                                float* __restrict__ v,
                                                float* __restrict__ out) {
    const int wave = threadIdx.x >> 5, lane = threadIdx.x & 31;
    const int item = blockIdx.x * 8 + wave;    // 0..511 = b*16+n
    const int idx  = item * DLEN + lane;
    float sv = s[idx];
    float ss = sv * sv;
#pragma unroll
    for (int off = 16; off > 0; off >>= 1) ss += __shfl_xor(ss, off, 32);
    float scale = (ss / (1.0f + ss)) * rsqrtf(ss);
    float r = scale * sv;
    v[idx] = r;
    out[idx] = r;
}

// bias[b,q,n] += dot(u[b,q,n,:], v[b,n,:]); wave per (b,q), v cached in LDS.
__global__ __launch_bounds__(256) void k_bias_update(const _Float16* __restrict__ u,
                                                     const float* __restrict__ v,
                                                     float* __restrict__ bias) {
    __shared__ __align__(16) float vl[OC];
    const int tid   = threadIdx.x;
    const int item0 = blockIdx.x * 8;
    const int b     = item0 / QTOT;            // constant per block (QTOT%8==0)
    vl[tid]       = v[(size_t)b * OC + tid];
    vl[tid + 256] = v[(size_t)b * OC + tid + 256];
    __syncthreads();

    const int wave = tid >> 5, lane = tid & 31;
    const int q    = (item0 + wave) - b * QTOT;
    const _Float16* up = u + ((size_t)b * QTOT + q) * OC + lane * 16;
    v8h u0 = *(const v8h*)up;                  // 16B
    v8h u1 = *(const v8h*)(up + 8);            // 16B
    const int n = lane >> 1;
    const float4* vp = (const float4*)&vl[n * DLEN + (lane & 1) * 16];
    float4 w0 = vp[0], w1 = vp[1], w2 = vp[2], w3 = vp[3];  // 4x ds_read_b128
    float sum = 0.0f;
    sum += (float)u0[0] * w0.x; sum += (float)u0[1] * w0.y;
    sum += (float)u0[2] * w0.z; sum += (float)u0[3] * w0.w;
    sum += (float)u0[4] * w1.x; sum += (float)u0[5] * w1.y;
    sum += (float)u0[6] * w1.z; sum += (float)u0[7] * w1.w;
    sum += (float)u1[0] * w2.x; sum += (float)u1[1] * w2.y;
    sum += (float)u1[2] * w2.z; sum += (float)u1[3] * w2.w;
    sum += (float)u1[4] * w3.x; sum += (float)u1[5] * w3.y;
    sum += (float)u1[6] * w3.z; sum += (float)u1[7] * w3.w;
    sum += __shfl_xor(sum, 1, 32);             // combine the two d-halves
    if ((lane & 1) == 0)
        bias[((size_t)b * QTOT + q) * NCAP + n] += sum;
}

// ---------------------------------------------------------------------------
extern "C" void kernel_launch(void* const* d_in, const int* in_sizes, int n_in,
                              void* d_out, int out_size, void* d_ws, size_t ws_size,
                              hipStream_t stream) {
    const float* x = (const float*)d_in[0];   // tensor (B,P,H,W,NPRIM)
    const float* w = (const float*)d_in[1];   // weight (Q,NPRIM,OC)
    float* out = (float*)d_out;               // v (B,NCAP,DLEN)

    char* ws = (char*)d_ws;
    size_t off = 0;
    _Float16* u  = (_Float16*)(ws + off); off += (size_t)BB * QTOT * OC * 2;   // 151 MB
    float* bias  = (float*)(ws + off);    off += (size_t)BB * QTOT * NCAP * 4; // 9.4 MB
    float* c     = (float*)(ws + off);    off += (size_t)BB * QTOT * NCAP * 4; // 9.4 MB
    float* sp    = (float*)(ws + off);    off += (size_t)NQC * BB * OC * 4;    // 2 MB
    float* s     = (float*)(ws + off);    off += (size_t)BB * OC * 4;
    float* v     = (float*)(ws + off);    off += (size_t)BB * OC * 4;
    (void)ws_size; (void)in_sizes; (void)n_in; (void)out_size;

    const int NBQ = BB * QTOT;                       // 147456
    k_zero<<<(NBQ * NCAP + 255) / 256, 256, 0, stream>>>(bias, NBQ * NCAP);
    k_compute_u<<<QTOT, 256, 0, stream>>>(x, w, u);

    for (int it = 0; it < 3; ++it) {
        k_softmax<<<(NBQ + 255) / 256, 256, 0, stream>>>(bias, c);
        k_route_s<<<256, 256, 0, stream>>>(u, c, sp);     // 2048 waves
        k_reduce_s<<<64, 256, 0, stream>>>(sp, s);
        k_squash<<<64, 256, 0, stream>>>(s, v, out);      // 512 waves
        if (it < 2)
            k_bias_update<<<NBQ / 8, 256, 0, stream>>>(u, v, bias);
    }
}
